// MoE_77421080477766
// MI455X (gfx1250) — matrix operand order repaired and checked by reading, not compiled, not run on hardware
//
#include <hip/hip_runtime.h>

// ---------------- problem constants ----------------
#define S_TOK   4096
#define C_DIM   1024
#define H_DIM   2744
#define NE      8
#define MAXROWS 13440          // 105 tiles * 128 rows (worst case segment padding)
#define MTILES  105

typedef __attribute__((ext_vector_type(16))) _Float16 v16h;
typedef __attribute__((ext_vector_type(8)))  float    v8f;

// ---------------- async global->LDS (gfx1250), guarded ----------------
#if defined(__HIP_DEVICE_COMPILE__) && __has_builtin(__builtin_amdgcn_global_load_async_to_lds_b128)
#define USE_ASYNC_LDS 1
typedef __attribute__((ext_vector_type(4))) int int4v;
typedef __attribute__((address_space(1))) int4v g_int4v;   // global int4*
typedef __attribute__((address_space(3))) int4v l_int4v;   // LDS int4*
// generic->AS1 / generic-shared->AS3 via integer round trip (LDS offset = low 32 bits)
#define TO_GLB(p) ((g_int4v*)(uintptr_t)(p))
#define TO_LDS(p) ((l_int4v*)(uintptr_t)(uint32_t)(uintptr_t)(p))
#else
#define USE_ASYNC_LDS 0
#endif

__device__ __forceinline__ void wait_async0() {
#if USE_ASYNC_LDS
#if __has_builtin(__builtin_amdgcn_s_wait_asynccnt)
  __builtin_amdgcn_s_wait_asynccnt(0);
#else
  asm volatile("s_wait_asynccnt 0" ::: "memory");
#endif
#endif
}

// ---------------- workspace layout (bytes) ----------------
#define WS_CNT     0        // int[16]   per-expert routed counts (cnt[8]=4096 shared)
#define WS_CNT2    64       // int[16]   scatter cursors
#define WS_IMP     128      // float[8]  importance accum
#define WS_LOAD    160      // float[8]  load accum
#define WS_SEG     192      // int[16]   aligned segment offsets (10 used)
#define WS_TKI     256      // int[2*4096]
#define WS_TKW     33024    // float[2*4096]
#define WS_ROWTOK  65792    // int[MAXROWS]
#define WS_ROWW    119552   // float[MAXROWS]
#define WS_H       173312   // _Float16[MAXROWS*H_DIM]  (16B aligned)

// ---------------- WMMA helpers ----------------
__device__ __forceinline__ v8f wmma_f16(v16h a, v16h b, v8f c) {
  // D = A(16x32 f16) * B(32x16 f16) + C(16x16 f32)
  return __builtin_amdgcn_wmma_f32_16x16x32_f16(false, a, false, b, (short)0, c,
                                                false, false);
}

// A fragment: lane L holds row (L&15); VGPR0-3 = K[8*(L>>4) .. +8),
// VGPR4-7 = K[16+8*(L>>4) .. +8).  LDS tile is [128][40] f16 (row-major K).
__device__ __forceinline__ v16h frag_a(const _Float16* sA, int row0, int lane) {
  const _Float16* p = sA + (row0 + (lane & 15)) * 40 + ((lane >> 4) << 3);
  union { v16h v; uint4 u[2]; } f;
  f.u[0] = *(const uint4*)(p);
  f.u[1] = *(const uint4*)(p + 16);
  return f.v;
}

// B fragment: lane L holds column n=nbase+(L&15); elems 0-7 = K[8*(L>>4) .. +8),
// elems 8-15 = K[16+8*(L>>4) .. +8).  LDS tile is transposed: [32 k][72 n] f16.
__device__ __forceinline__ v16h frag_b(const _Float16* sBt, int nbase, int lane) {
  const _Float16* p = sBt + ((lane >> 4) << 3) * 72 + nbase + (lane & 15);
  v16h v;
#pragma unroll
  for (int s = 0; s < 8; ++s) { v[s] = p[s * 72]; v[8 + s] = p[(16 + s) * 72]; }
  return v;
}

// ---------------- kernel 0: init ----------------
__global__ __launch_bounds__(256) void moe_init(float* __restrict__ y,
    int* __restrict__ cnt, int* __restrict__ cnt2,
    float* __restrict__ imp, float* __restrict__ loadv,
    int* __restrict__ rowtok, float* __restrict__ roww) {
  const int idx = blockIdx.x * blockDim.x + threadIdx.x;
  const int stride = gridDim.x * blockDim.x;
  for (int i = idx; i < S_TOK * C_DIM; i += stride) y[i] = 0.f;
  if (idx < 16) { cnt[idx] = (idx == NE) ? S_TOK : 0; cnt2[idx] = 0; }
  if (idx < NE) { imp[idx] = 0.f; loadv[idx] = 0.f; }
  if (idx < MAXROWS) { rowtok[idx] = -1; roww[idx] = 0.f; }
}

// ---------------- kernel 1: gating (wave per token) ----------------
__global__ __launch_bounds__(256) void moe_gate(const float* __restrict__ x,
    const float* __restrict__ gw, int* __restrict__ tki, float* __restrict__ tkw,
    int* __restrict__ cnt, float* __restrict__ imp, float* __restrict__ loadv) {
  const int lane = threadIdx.x & 31;
  const int wid  = threadIdx.x >> 5;
  const int s    = blockIdx.x * 8 + wid;
  const float* xr = x + (size_t)s * C_DIM;

  float acc[NE];
#pragma unroll
  for (int e = 0; e < NE; ++e) acc[e] = 0.f;
  for (int i = 0; i < 32; ++i) {
    const float xv = xr[i * 32 + lane];
#pragma unroll
    for (int e = 0; e < NE; ++e) acc[e] += xv * gw[e * C_DIM + i * 32 + lane];
  }
#pragma unroll
  for (int e = 0; e < NE; ++e)
    for (int off = 16; off >= 1; off >>= 1) acc[e] += __shfl_xor(acc[e], off, 32);

  // softmax (f32)
  float mx = acc[0];
#pragma unroll
  for (int e = 1; e < NE; ++e) mx = fmaxf(mx, acc[e]);
  float p[NE], den = 0.f;
#pragma unroll
  for (int e = 0; e < NE; ++e) { p[e] = __expf(acc[e] - mx); den += p[e]; }
  const float inv = 1.f / den;
#pragma unroll
  for (int e = 0; e < NE; ++e) p[e] *= inv;

  // top-2 (ties -> lowest index, matching lax.top_k)
  int i0 = 0; float v0 = p[0];
#pragma unroll
  for (int e = 1; e < NE; ++e) if (p[e] > v0) { v0 = p[e]; i0 = e; }
  int i1 = -1; float v1 = -1.f;
#pragma unroll
  for (int e = 0; e < NE; ++e) if (e != i0 && p[e] > v1) { v1 = p[e]; i1 = e; }

  if (lane == 0) {
    const float wsum = v0 + v1;
    tki[s] = i0; tki[S_TOK + s] = i1;
    tkw[s] = v0 / wsum; tkw[S_TOK + s] = v1 / wsum;
    atomicAdd(&cnt[i0], 1); atomicAdd(&cnt[i1], 1);
#pragma unroll
    for (int e = 0; e < NE; ++e) atomicAdd(&imp[e], p[e] * (1.f / S_TOK));
    atomicAdd(&loadv[i0], 1.f / S_TOK);
  }
}

// ---------------- kernel 2: segment scan + aux loss ----------------
__global__ void moe_scan(const int* __restrict__ cnt, int* __restrict__ seg,
                         const float* __restrict__ imp, const float* __restrict__ loadv,
                         float* __restrict__ aux_out) {
  if (threadIdx.x == 0) {
    int off = 0;
    for (int e = 0; e < NE + 1; ++e) {
      seg[e] = off;
      off += ((cnt[e] + 127) >> 7) << 7;   // 128-row aligned segments
    }
    seg[NE + 1] = off;
    float a = 0.f;
    for (int e = 0; e < NE; ++e) a += imp[e] * loadv[e];
    aux_out[0] = (float)NE * a;
  }
}

// ---------------- kernel 3: scatter tokens to segments ----------------
__global__ __launch_bounds__(256) void moe_scatter(const int* __restrict__ tki,
    const float* __restrict__ tkw, const int* __restrict__ seg,
    int* __restrict__ cnt2, int* __restrict__ rowtok, float* __restrict__ roww) {
  const int s = blockIdx.x * blockDim.x + threadIdx.x;
  if (s >= S_TOK) return;
#pragma unroll
  for (int k = 0; k < 2; ++k) {
    const int e = tki[k * S_TOK + s];
    const float w = tkw[k * S_TOK + s];
    const int p = seg[e] + atomicAdd(&cnt2[e], 1);
    rowtok[p] = s; roww[p] = w;
  }
  const int p = seg[NE] + s;  // shared pseudo-expert: every token, weight 1
  rowtok[p] = s; roww[p] = 1.f;
}

// ---------------- kernel 4: grouped GEMM1  h = silu(x@w1^T) * (x@w2^T) ----------------
// tile M=128 rows, N=64 hidden cols, K=1024. 8 waves: 4(M) x 2(N), each 32x32.
// Double-buffered LDS: one barrier per K-step, stage(k+1) overlaps wmma(k).
__global__ __launch_bounds__(256) void moe_gemm1(const float* __restrict__ x,
    const float* __restrict__ w1, const float* __restrict__ w2,
    const float* __restrict__ sw1, const float* __restrict__ sw2,
    const int* __restrict__ seg, const int* __restrict__ cnt,
    const int* __restrict__ rowtok, _Float16* __restrict__ hbuf) {
  const int m0 = blockIdx.x * 128;
  int e = -1;
  for (int i = 0; i < NE + 1; ++i)
    if (m0 >= seg[i] && m0 < seg[i + 1]) { e = i; break; }
  if (e < 0) return;
  if (m0 >= seg[e] + cnt[e]) return;  // fully-pad tile

  const float* bp1 = (e < NE) ? (w1 + (size_t)e * H_DIM * C_DIM) : sw1;
  const float* bp2 = (e < NE) ? (w2 + (size_t)e * H_DIM * C_DIM) : sw2;
  const int n0 = blockIdx.y * 64;

  __shared__ _Float16 sA[2][128 * 40];
  __shared__ _Float16 sB1[2][32 * 72];
  __shared__ _Float16 sB2[2][32 * 72];

  const int tid  = threadIdx.x;
  const int lane = tid & 31;
  const int wid  = tid >> 5;
  const int wm   = (wid >> 1) * 32;
  const int wn   = (wid & 1) * 32;

  // staging roles
  const int ar = tid >> 1;            // A row 0..127
  const int ak = (tid & 1) * 16;      // A K-suboffset 0/16
  const int tok = rowtok[m0 + ar];
  const int srow = (tok < 0) ? 0 : tok;
  const int bn   = tid >> 2;          // B n index 0..63 (4 lanes share a row -> coalesced)
  const int koff = (tid & 3) * 8;     // B K-suboffset 0/8/16/24
  const bool bvalid = (n0 + bn) < H_DIM;

  auto stageA = [&](int k0, _Float16* dA) {
    const float* gp = x + (size_t)srow * C_DIM + k0 + ak;
#pragma unroll
    for (int j = 0; j < 4; ++j) {
      const float4 v = *(const float4*)(gp + j * 4);
      union { _Float16 h[4]; uint2 u; } t;
      t.h[0] = (_Float16)v.x; t.h[1] = (_Float16)v.y;
      t.h[2] = (_Float16)v.z; t.h[3] = (_Float16)v.w;
      *(uint2*)&dA[ar * 40 + ak + j * 4] = t.u;
    }
    if (k0 + 32 < C_DIM) __builtin_prefetch(gp + 32, 0, 3);
  };
  auto stageB = [&](int k0, _Float16* d1, _Float16* d2) {
    const float* p1 = bp1 + (size_t)(n0 + bn) * C_DIM + k0 + koff;
    const float* p2 = bp2 + (size_t)(n0 + bn) * C_DIM + k0 + koff;
#pragma unroll
    for (int hh = 0; hh < 2; ++hh) {
      float4 v1, v2;
      if (bvalid) {
        v1 = *(const float4*)(p1 + hh * 4);
        v2 = *(const float4*)(p2 + hh * 4);
      } else {
        v1 = make_float4(0.f, 0.f, 0.f, 0.f); v2 = v1;
      }
      const int kb = koff + hh * 4;
      d1[(kb + 0) * 72 + bn] = (_Float16)v1.x;
      d1[(kb + 1) * 72 + bn] = (_Float16)v1.y;
      d1[(kb + 2) * 72 + bn] = (_Float16)v1.z;
      d1[(kb + 3) * 72 + bn] = (_Float16)v1.w;
      d2[(kb + 0) * 72 + bn] = (_Float16)v2.x;
      d2[(kb + 1) * 72 + bn] = (_Float16)v2.y;
      d2[(kb + 2) * 72 + bn] = (_Float16)v2.z;
      d2[(kb + 3) * 72 + bn] = (_Float16)v2.w;
    }
    if (bvalid && k0 + 32 < C_DIM) {
      __builtin_prefetch(p1 + 32, 0, 3);
      __builtin_prefetch(p2 + 32, 0, 3);
    }
  };

  v8f acc1[2][2] = {};
  v8f acc2[2][2] = {};

  stageA(0, sA[0]); stageB(0, sB1[0], sB2[0]);
  const int NK = C_DIM / 32;
  for (int kt = 0; kt < NK; ++kt) {
    const int cur = kt & 1, nxt = cur ^ 1;
    __syncthreads();
    if (kt + 1 < NK) { stageA((kt + 1) * 32, sA[nxt]); stageB((kt + 1) * 32, sB1[nxt], sB2[nxt]); }

    const v16h a0 = frag_a(sA[cur], wm + 0,  lane);
    const v16h a1 = frag_a(sA[cur], wm + 16, lane);
#pragma unroll
    for (int nb = 0; nb < 2; ++nb) {
      const v16h b1 = frag_b(sB1[cur], wn + nb * 16, lane);
      acc1[0][nb] = wmma_f16(a0, b1, acc1[0][nb]);
      acc1[1][nb] = wmma_f16(a1, b1, acc1[1][nb]);
      const v16h b2 = frag_b(sB2[cur], wn + nb * 16, lane);
      acc2[0][nb] = wmma_f16(a0, b2, acc2[0][nb]);
      acc2[1][nb] = wmma_f16(a1, b2, acc2[1][nb]);
    }
  }

  // epilogue: h = silu(a1)*a2, store f16
#pragma unroll
  for (int mb = 0; mb < 2; ++mb)
#pragma unroll
    for (int nb = 0; nb < 2; ++nb)
#pragma unroll
      for (int v = 0; v < 8; ++v) {
        const int row = m0 + wm + mb * 16 + ((lane >> 4) << 3) + v;
        const int col = n0 + wn + nb * 16 + (lane & 15);
        if (col < H_DIM) {
          const float a = acc1[mb][nb][v];
          const float b = acc2[mb][nb][v];
          const float hv = (a / (1.f + __expf(-a))) * b;
          hbuf[(size_t)row * H_DIM + col] = (_Float16)hv;
        }
      }
}

// ---------------- kernel 5: grouped GEMM2  y += wgt * (h @ w3^T) ----------------
// tile M=128 rows, N=64 out cols, K=2744.  A-operand (f16) staged via
// GLOBAL_LOAD_ASYNC_TO_LDS_B128 when available (ASYNCcnt pipeline).
__global__ __launch_bounds__(256) void moe_gemm2(const _Float16* __restrict__ hbuf,
    const float* __restrict__ w3, const float* __restrict__ sw3,
    const int* __restrict__ seg, const int* __restrict__ cnt,
    const int* __restrict__ rowtok, const float* __restrict__ roww,
    float* __restrict__ y) {
  const int m0 = blockIdx.x * 128;
  int e = -1;
  for (int i = 0; i < NE + 1; ++i)
    if (m0 >= seg[i] && m0 < seg[i + 1]) { e = i; break; }
  if (e < 0) return;
  if (m0 >= seg[e] + cnt[e]) return;

  const float* bp3 = (e < NE) ? (w3 + (size_t)e * C_DIM * H_DIM) : sw3;
  const int n0 = blockIdx.y * 64;

  __shared__ _Float16 sA[2][128 * 40];
  __shared__ _Float16 sBt[2][32 * 72];
  __shared__ int   stok[128];
  __shared__ float swgt[128];

  const int tid  = threadIdx.x;
  const int lane = tid & 31;
  const int wid  = tid >> 5;
  const int wm   = (wid >> 1) * 32;
  const int wn   = (wid & 1) * 32;

  const int ar = tid >> 1;
  const int ak = (tid & 1) * 16;
  const int bn   = tid >> 2;
  const int koff = (tid & 3) * 8;

  if (tid < 128) { stok[tid] = rowtok[m0 + tid]; swgt[tid] = roww[m0 + tid]; }

  const _Float16* arow = hbuf + (size_t)(m0 + ar) * H_DIM;
  const float*    brow = bp3 + (size_t)(n0 + bn) * H_DIM;

  auto stageA2 = [&](int k0, _Float16* dA) {
    if (k0 + 32 <= H_DIM) {
#if USE_ASYNC_LDS
      __builtin_amdgcn_global_load_async_to_lds_b128(
          TO_GLB(arow + k0 + ak), TO_LDS(&dA[ar * 40 + ak]), 0, 0);
      __builtin_amdgcn_global_load_async_to_lds_b128(
          TO_GLB(arow + k0 + ak + 8), TO_LDS(&dA[ar * 40 + ak + 8]), 0, 0);
#else
      *(uint4*)&dA[ar * 40 + ak + 0] = *(const uint4*)(arow + k0 + ak + 0);
      *(uint4*)&dA[ar * 40 + ak + 8] = *(const uint4*)(arow + k0 + ak + 8);
#endif
      if (k0 + 64 <= H_DIM) __builtin_prefetch(arow + k0 + ak + 32, 0, 3);
    } else {  // K tail (2744 % 32 == 24): elementwise zero-filled
#pragma unroll
      for (int j = 0; j < 16; ++j) {
        const int k = k0 + ak + j;
        dA[ar * 40 + ak + j] = (k < H_DIM) ? arow[k] : (_Float16)0.f;
      }
    }
  };
  auto stageB2 = [&](int k0, _Float16* dB) {
    if (k0 + 32 <= H_DIM) {
#pragma unroll
      for (int hh = 0; hh < 2; ++hh) {
        const float4 v = *(const float4*)(brow + k0 + koff + hh * 4);
        const int kb = koff + hh * 4;
        dB[(kb + 0) * 72 + bn] = (_Float16)v.x;
        dB[(kb + 1) * 72 + bn] = (_Float16)v.y;
        dB[(kb + 2) * 72 + bn] = (_Float16)v.z;
        dB[(kb + 3) * 72 + bn] = (_Float16)v.w;
      }
      if (k0 + 64 <= H_DIM) __builtin_prefetch(brow + k0 + koff + 32, 0, 3);
    } else {
#pragma unroll
      for (int hh = 0; hh < 2; ++hh)
#pragma unroll
        for (int u = 0; u < 4; ++u) {
          const int kb = koff + hh * 4 + u;
          const int k = k0 + kb;
          dB[kb * 72 + bn] = (k < H_DIM) ? (_Float16)brow[k] : (_Float16)0.f;
        }
    }
  };

  v8f acc[2][2] = {};

  stageA2(0, sA[0]); stageB2(0, sBt[0]);
  const int NK = (H_DIM + 31) / 32;  // 86
  for (int kt = 0; kt < NK; ++kt) {
    const int cur = kt & 1, nxt = cur ^ 1;
    wait_async0();          // our wave's async copies into LDS are done
    __syncthreads();        // everyone's are done
    if (kt + 1 < NK) { stageA2((kt + 1) * 32, sA[nxt]); stageB2((kt + 1) * 32, sBt[nxt]); }

    const v16h a0 = frag_a(sA[cur], wm + 0,  lane);
    const v16h a1 = frag_a(sA[cur], wm + 16, lane);
#pragma unroll
    for (int nb = 0; nb < 2; ++nb) {
      const v16h b = frag_b(sBt[cur], wn + nb * 16, lane);
      acc[0][nb] = wmma_f16(a0, b, acc[0][nb]);
      acc[1][nb] = wmma_f16(a1, b, acc[1][nb]);
    }
  }

  // epilogue: scale by gate weight, atomically accumulate into y
#pragma unroll
  for (int mb = 0; mb < 2; ++mb)
#pragma unroll
    for (int nb = 0; nb < 2; ++nb)
#pragma unroll
      for (int v = 0; v < 8; ++v) {
        const int rl = wm + mb * 16 + ((lane >> 4) << 3) + v;
        const int t2 = stok[rl];
        if (t2 >= 0) {
          const float val = acc[mb][nb][v] * swgt[rl];
          const int col = n0 + wn + nb * 16 + (lane & 15);
          atomicAdd(&y[(size_t)t2 * C_DIM + col], val);
        }
      }
}

// ---------------- launcher ----------------
extern "C" void kernel_launch(void* const* d_in, const int* in_sizes, int n_in,
                              void* d_out, int out_size, void* d_ws, size_t ws_size,
                              hipStream_t stream) {
  const float* x      = (const float*)d_in[0];
  const float* gate_w = (const float*)d_in[1];
  const float* w1     = (const float*)d_in[2];
  const float* w2     = (const float*)d_in[3];
  const float* w3     = (const float*)d_in[4];
  const float* sw1    = (const float*)d_in[5];
  const float* sw2    = (const float*)d_in[6];
  const float* sw3    = (const float*)d_in[7];
  float* y = (float*)d_out;

  char* ws = (char*)d_ws;
  int*   cnt    = (int*)(ws + WS_CNT);
  int*   cnt2   = (int*)(ws + WS_CNT2);
  float* imp    = (float*)(ws + WS_IMP);
  float* loadv  = (float*)(ws + WS_LOAD);
  int*   seg    = (int*)(ws + WS_SEG);
  int*   tki    = (int*)(ws + WS_TKI);
  float* tkw    = (float*)(ws + WS_TKW);
  int*   rowtok = (int*)(ws + WS_ROWTOK);
  float* roww   = (float*)(ws + WS_ROWW);
  _Float16* hbuf = (_Float16*)(ws + WS_H);

  moe_init<<<4096, 256, 0, stream>>>(y, cnt, cnt2, imp, loadv, rowtok, roww);
  moe_gate<<<S_TOK / 8, 256, 0, stream>>>(x, gate_w, tki, tkw, cnt, imp, loadv);
  moe_scan<<<1, 32, 0, stream>>>(cnt, seg, imp, loadv, y + (size_t)S_TOK * C_DIM);
  moe_scatter<<<S_TOK / 256, 256, 0, stream>>>(tki, tkw, seg, cnt2, rowtok, roww);
  moe_gemm1<<<dim3(MTILES, (H_DIM + 63) / 64), 256, 0, stream>>>(
      x, w1, w2, sw1, sw2, seg, cnt, rowtok, hbuf);
  moe_gemm2<<<dim3(MTILES, C_DIM / 64), 256, 0, stream>>>(
      hbuf, w3, sw3, seg, cnt, rowtok, roww, y);
}
